// EncoderLayer_5351529250947
// MI455X (gfx1250) — compile-verified, matrix operand order checked by reading, not run on hardware
//
#include <hip/hip_runtime.h>

#define S_LEN   2048
#define DM      512
#define N_HEAD  8
#define D_HEAD  64
#define FFN_DIM 2048
#define BATCH   2
#define ROWS    (BATCH * S_LEN) // 4096

typedef __bf16 bf16x8  __attribute__((ext_vector_type(8)));
typedef __bf16 bf16x16 __attribute__((ext_vector_type(16)));
typedef float  f32x8   __attribute__((ext_vector_type(8)));

__device__ __forceinline__ f32x8 wmma_bf16(bf16x16 a, bf16x16 b, f32x8 c) {
  return __builtin_amdgcn_wmma_f32_16x16x32_bf16(false, a, false, b, (short)0, c, false, false);
}

// Load one 16x32 operand chunk in the CDNA5 16-bit A/B layout from a row-major
// source (row r holds K contiguous). Lanes 0-15: K = kbase+0..7 and +16..23;
// lanes 16-31: K = kbase+8..15 and +24..31. (8 VGPRs, K-pairs packed.)
__device__ __forceinline__ bf16x16 load_tile(const __bf16* __restrict__ base, int row,
                                             int ld, int kbase, int half) {
  const __bf16* p = base + (size_t)row * ld + kbase + half * 8;
  bf16x8 lo = *(const bf16x8*)(p);
  bf16x8 hi = *(const bf16x8*)(p + 16);
  return __builtin_shufflevector(lo, hi, 0,1,2,3,4,5,6,7,8,9,10,11,12,13,14,15);
}

__device__ __forceinline__ float lane_shfl(float v, int srcLane) {
  return __int_as_float(__builtin_amdgcn_ds_bpermute(srcLane << 2, __float_as_int(v)));
}
__device__ __forceinline__ float shfl_xor_f(float v, int mask) {
  return lane_shfl(v, ((int)(threadIdx.x & 31)) ^ mask);
}

// ---------------------------------------------------------------- convert/transpose
__global__ __launch_bounds__(256) void cvt_kernel(const float* __restrict__ in,
                                                  __bf16* __restrict__ out,
                                                  int rows, int cols, int transpose) {
  int idx = blockIdx.x * 256 + threadIdx.x;
  int n = rows * cols;
  if (idx >= n) return;
  int r = idx / cols, c = idx - r * cols;
  float v = in[idx];
  out[transpose ? ((size_t)c * rows + r) : (size_t)idx] = (__bf16)v;
}

// ---------------------------------------------------------------- generic WMMA GEMM
// C[M,N] = A[M,K] @ BT[N,K]^T (+bias). Wave computes 32x64 (2 M-subtiles x 4
// N-subtiles). K loop is ping-pong double-buffered with step 64: loads for the
// next chunk are written directly into the loop-carried operand registers, so
// no rotation copies; compiler emits partial s_wait_loadcnt. Requires K % 64
// == 0 and K >= 128 (true for 512 / 2048). Epilogue by mode:
// 0: q  -> bf16 [b,h,s,d], *0.125 (1/sqrt(D_HEAD))
// 1: k  -> bf16 [b,h,s,d]
// 2: v  -> bf16 [b,h,d,s] (transposed for PV B-tiles)
// 3: relu -> bf16 row-major
// 4: f32 row-major
__global__ __launch_bounds__(128) void gemm_kernel(const __bf16* __restrict__ A,
                                                   const __bf16* __restrict__ BT,
                                                   const float* __restrict__ bias,
                                                   void* __restrict__ out,
                                                   int M, int N, int K, int mode) {
  int wid = threadIdx.x >> 5;
  int ln  = threadIdx.x & 31;
  int half = ln >> 4, lr = ln & 15;
  int ntn  = N >> 6;
  int tile = blockIdx.x * 4 + wid;
  int m0 = (tile / ntn) * 32;
  int n0 = (tile % ntn) * 64;
  if (m0 >= M) return;

  f32x8 acc[2][4] = {};

  // S0 buffer: chunk kc (loop-carried)
  bf16x16 a0 = load_tile(A, m0 + lr,      K, 0, half);
  bf16x16 a1 = load_tile(A, m0 + 16 + lr, K, 0, half);
  bf16x16 bb[4];
#pragma unroll
  for (int ns = 0; ns < 4; ++ns) bb[ns] = load_tile(BT, n0 + 16 * ns + lr, K, 0, half);

  int kc = 0;
#pragma unroll 1
  for (; kc + 64 < K; kc += 64) {
    // S1 buffer: chunk kc+32 (fresh temps, no loop carry)
    bf16x16 c0 = load_tile(A, m0 + lr,      K, kc + 32, half);
    bf16x16 c1 = load_tile(A, m0 + 16 + lr, K, kc + 32, half);
    bf16x16 cb[4];
#pragma unroll
    for (int ns = 0; ns < 4; ++ns) cb[ns] = load_tile(BT, n0 + 16 * ns + lr, K, kc + 32, half);
    // consume S0 (chunk kc)
#pragma unroll
    for (int ns = 0; ns < 4; ++ns) {
      acc[0][ns] = wmma_bf16(a0, bb[ns], acc[0][ns]);
      acc[1][ns] = wmma_bf16(a1, bb[ns], acc[1][ns]);
    }
    // refill S0 directly (chunk kc+64) - loads land in the loop-carried regs
    a0 = load_tile(A, m0 + lr,      K, kc + 64, half);
    a1 = load_tile(A, m0 + 16 + lr, K, kc + 64, half);
#pragma unroll
    for (int ns = 0; ns < 4; ++ns) bb[ns] = load_tile(BT, n0 + 16 * ns + lr, K, kc + 64, half);
    // consume S1 (chunk kc+32)
#pragma unroll
    for (int ns = 0; ns < 4; ++ns) {
      acc[0][ns] = wmma_bf16(c0, cb[ns], acc[0][ns]);
      acc[1][ns] = wmma_bf16(c1, cb[ns], acc[1][ns]);
    }
  }
  // tail: chunks kc and kc+32 (kc + 64 == K)
  {
    bf16x16 c0 = load_tile(A, m0 + lr,      K, kc + 32, half);
    bf16x16 c1 = load_tile(A, m0 + 16 + lr, K, kc + 32, half);
    bf16x16 cb[4];
#pragma unroll
    for (int ns = 0; ns < 4; ++ns) cb[ns] = load_tile(BT, n0 + 16 * ns + lr, K, kc + 32, half);
#pragma unroll
    for (int ns = 0; ns < 4; ++ns) {
      acc[0][ns] = wmma_bf16(a0, bb[ns], acc[0][ns]);
      acc[1][ns] = wmma_bf16(a1, bb[ns], acc[1][ns]);
    }
#pragma unroll
    for (int ns = 0; ns < 4; ++ns) {
      acc[0][ns] = wmma_bf16(c0, cb[ns], acc[0][ns]);
      acc[1][ns] = wmma_bf16(c1, cb[ns], acc[1][ns]);
    }
  }

#pragma unroll
  for (int ms = 0; ms < 2; ++ms) {
#pragma unroll
    for (int ns = 0; ns < 4; ++ns) {
      int n = n0 + 16 * ns + lr;
      float bv = bias[n];
#pragma unroll
      for (int g = 0; g < 8; ++g) {
        int m = m0 + 16 * ms + g + 8 * half;   // C-layout row
        float v = acc[ms][ns][g] + bv;
        if (mode <= 2) {
          int b = m >> 11, s = m & (S_LEN - 1);
          int h = n >> 6,  d = n & 63;
          if (mode == 2) {
            ((__bf16*)out)[(((size_t)(b * N_HEAD + h) * D_HEAD + d) << 11) + s] = (__bf16)v;
          } else {
            float sv = (mode == 0) ? v * 0.125f : v;
            ((__bf16*)out)[(((size_t)(b * N_HEAD + h) * S_LEN + s) << 6) + d] = (__bf16)sv;
          }
        } else if (mode == 3) {
          ((__bf16*)out)[(size_t)m * N + n] = (__bf16)fmaxf(v, 0.0f);
        } else {
          ((float*)out)[(size_t)m * N + n] = v;
        }
      }
    }
  }
}

// ---------------------------------------------------------------- attention (flash + skew)
// q,k: bf16 [B*H][S][64] (q pre-scaled 1/8), vt: bf16 [B*H][64][S], er: bf16 [S][64]
// ctx: f32 [B][S][512]. One wave = 16 query rows; key blocks of 32, causal.
__global__ __launch_bounds__(128) void attn_kernel(const __bf16* __restrict__ q,
                                                   const __bf16* __restrict__ k,
                                                   const __bf16* __restrict__ vt,
                                                   const __bf16* __restrict__ er,
                                                   float* __restrict__ ctx) {
  int wid = threadIdx.x >> 5;
  int ln  = threadIdx.x & 31;
  int half = ln >> 4, lr = ln & 15;
  int bh = blockIdx.y;
  int b  = bh >> 3, h = bh & 7;
  int i0 = (blockIdx.x * 4 + wid) * 16;

  const __bf16* qp = q  + ((size_t)bh << 17);
  const __bf16* kp = k  + ((size_t)bh << 17);
  const __bf16* vp = vt + ((size_t)bh << 17);

  __shared__ float pbuf[4][16][32];
  float (*pb)[32] = pbuf[wid];

  bf16x16 qa0 = load_tile(qp, i0 + lr, D_HEAD, 0, half);
  bf16x16 qa1 = load_tile(qp, i0 + lr, D_HEAD, 32, half);

  f32x8 acc[4] = {};
  float rm[8], rl[8];
#pragma unroll
  for (int g = 0; g < 8; ++g) { rm[g] = -1e30f; rl[g] = 0.0f; }

  int nblk = (i0 + 15) / 32 + 1;                 // wave-uniform
  for (int bk = 0; bk < nblk; ++bk) {
    int j0 = bk * 32;
    // ---- QK^T : 16x32 scores (2 subtiles), K=64 in two chunks
    f32x8 s0 = {}, s1 = {};
    {
      bf16x16 kb0 = load_tile(kp, j0 + lr,      D_HEAD, 0,  half);
      bf16x16 kb1 = load_tile(kp, j0 + lr,      D_HEAD, 32, half);
      bf16x16 kb2 = load_tile(kp, j0 + 16 + lr, D_HEAD, 0,  half);
      bf16x16 kb3 = load_tile(kp, j0 + 16 + lr, D_HEAD, 32, half);
      s0 = wmma_bf16(qa0, kb0, s0);
      s0 = wmma_bf16(qa1, kb1, s0);
      s1 = wmma_bf16(qa0, kb2, s1);
      s1 = wmma_bf16(qa1, kb3, s1);
    }
    // ---- relative-position band: Band[a,c] = q[i0+a] . Er[r0-16+c], c in [0,48)
    int r0 = j0 - i0 + (S_LEN - 1);
    f32x8 band[3];
#pragma unroll
    for (int t = 0; t < 3; ++t) {
      int l = r0 - 16 + 16 * t + lr;
      l = l < 0 ? 0 : (l > S_LEN - 1 ? S_LEN - 1 : l);   // clamp (masked cells only)
      bf16x16 eb0 = load_tile(er, l, D_HEAD, 0,  half);
      bf16x16 eb1 = load_tile(er, l, D_HEAD, 32, half);
      f32x8 bb = {};
      bb = wmma_bf16(qa0, eb0, bb);
      bb = wmma_bf16(qa1, eb1, bb);
      band[t] = bb;
    }
    // ---- add skewed Srel, causal mask, online softmax
#pragma unroll
    for (int g = 0; g < 8; ++g) {
      int a = g + 8 * half;                      // query row within tile
      int srcl = 16 * half + ((lr - a) & 15);    // shifted-diagonal gather lane
      float blo = lane_shfl(band[0][g], srcl);
      float bmd = lane_shfl(band[1][g], srcl);
      float bhi = lane_shfl(band[2][g], srcl);
      bool ge = (lr >= a);
      float srel0 = ge ? bmd : blo;              // cols j0 + lr
      float srel1 = ge ? bhi : bmd;              // cols j0 + 16 + lr
      int i = i0 + a;
      float sc0 = s0[g] + srel0;
      float sc1 = s1[g] + srel1;
      sc0 = (j0 + lr > i)      ? -1e30f : sc0;
      sc1 = (j0 + 16 + lr > i) ? -1e30f : sc1;
      float mx = fmaxf(sc0, sc1);
      mx = fmaxf(mx, shfl_xor_f(mx, 1));
      mx = fmaxf(mx, shfl_xor_f(mx, 2));
      mx = fmaxf(mx, shfl_xor_f(mx, 4));
      mx = fmaxf(mx, shfl_xor_f(mx, 8));
      float mnew  = fmaxf(rm[g], mx);
      float scale = __expf(rm[g] - mnew);
      rm[g] = mnew;
      float e0 = __expf(sc0 - mnew);
      float e1 = __expf(sc1 - mnew);
      float rs = e0 + e1;
      rs += shfl_xor_f(rs, 1);
      rs += shfl_xor_f(rs, 2);
      rs += shfl_xor_f(rs, 4);
      rs += shfl_xor_f(rs, 8);
      rl[g] = rl[g] * scale + rs;
      acc[0][g] *= scale; acc[1][g] *= scale; acc[2][g] *= scale; acc[3][g] *= scale;
      pb[a][lr]      = e0;                       // stage probs (C layout)
      pb[a][16 + lr] = e1;
    }
    // ---- probs C-layout -> A-layout via per-wave LDS (DS ops in-order per wave)
    float4 f0 = *(const float4*)&pb[lr][8 * half];
    float4 f1 = *(const float4*)&pb[lr][8 * half + 4];
    float4 f2 = *(const float4*)&pb[lr][16 + 8 * half];
    float4 f3 = *(const float4*)&pb[lr][16 + 8 * half + 4];
    bf16x16 pa;
    pa[0]=(__bf16)f0.x;  pa[1]=(__bf16)f0.y;  pa[2]=(__bf16)f0.z;  pa[3]=(__bf16)f0.w;
    pa[4]=(__bf16)f1.x;  pa[5]=(__bf16)f1.y;  pa[6]=(__bf16)f1.z;  pa[7]=(__bf16)f1.w;
    pa[8]=(__bf16)f2.x;  pa[9]=(__bf16)f2.y;  pa[10]=(__bf16)f2.z; pa[11]=(__bf16)f2.w;
    pa[12]=(__bf16)f3.x; pa[13]=(__bf16)f3.y; pa[14]=(__bf16)f3.z; pa[15]=(__bf16)f3.w;
    // ---- PV: out[16x64] += P[16x32] @ V[32x64] (V pre-transposed -> contiguous B tiles)
#pragma unroll
    for (int ds = 0; ds < 4; ++ds) {
      bf16x16 vb = load_tile(vp, 16 * ds + lr, S_LEN, j0, half);
      acc[ds] = wmma_bf16(pa, vb, acc[ds]);
    }
  }
  // ---- normalize + store ctx [b, s, h*64+d]
#pragma unroll
  for (int g = 0; g < 8; ++g) {
    float inv = 1.0f / rl[g];
    int row = i0 + g + 8 * half;
    float* op = ctx + ((size_t)b * S_LEN + row) * DM + h * D_HEAD;
#pragma unroll
    for (int ds = 0; ds < 4; ++ds) op[16 * ds + lr] = acc[ds][g] * inv;
  }
}

// ---------------------------------------------------------------- residual + LayerNorm
__global__ __launch_bounds__(256) void ln_kernel(const float* __restrict__ xa,
                                                 const float* __restrict__ xb,
                                                 const float* __restrict__ gamma,
                                                 const float* __restrict__ beta,
                                                 float* __restrict__ outf,
                                                 __bf16* __restrict__ outb) {
  int row = blockIdx.x;
  int t = threadIdx.x;
  const float* pa = xa + (size_t)row * DM;
  const float* pb = xb + (size_t)row * DM;
  float v0 = pa[t] + pb[t];
  float v1 = pa[t + 256] + pb[t + 256];
  float s = v0 + v1;
  float q = v0 * v0 + v1 * v1;
#pragma unroll
  for (int m = 1; m < 32; m <<= 1) { s += shfl_xor_f(s, m); q += shfl_xor_f(q, m); }
  __shared__ float ws[8], wq[8];
  if ((t & 31) == 0) { ws[t >> 5] = s; wq[t >> 5] = q; }
  __syncthreads();
  float S = 0.0f, Q = 0.0f;
#pragma unroll
  for (int i = 0; i < 8; ++i) { S += ws[i]; Q += wq[i]; }
  float mean = S * (1.0f / DM);
  float var  = Q * (1.0f / DM) - mean * mean;
  float rstd = rsqrtf(var + 1e-5f);
  float o0 = gamma[t]       * (v0 - mean) * rstd + beta[t];
  float o1 = gamma[t + 256] * (v1 - mean) * rstd + beta[t + 256];
  float* po = outf + (size_t)row * DM;
  po[t] = o0; po[t + 256] = o1;
  if (outb) {
    __bf16* pob = outb + (size_t)row * DM;
    pob[t] = (__bf16)o0; pob[t + 256] = (__bf16)o1;
  }
}

// ---------------------------------------------------------------- host launcher
extern "C" void kernel_launch(void* const* d_in, const int* in_sizes, int n_in,
                              void* d_out, int out_size, void* d_ws, size_t ws_size,
                              hipStream_t stream) {
  (void)in_sizes; (void)n_in; (void)out_size; (void)ws_size;
  const float* x   = (const float*)d_in[0];
  const float* Wq  = (const float*)d_in[1];
  const float* bq  = (const float*)d_in[2];
  const float* Wk  = (const float*)d_in[3];
  const float* bk  = (const float*)d_in[4];
  const float* Wv  = (const float*)d_in[5];
  const float* bv  = (const float*)d_in[6];
  const float* Er  = (const float*)d_in[7];
  const float* W1  = (const float*)d_in[8];
  const float* b1  = (const float*)d_in[9];
  const float* W2  = (const float*)d_in[10];
  const float* b2  = (const float*)d_in[11];
  const float* g1  = (const float*)d_in[12];
  const float* be1 = (const float*)d_in[13];
  const float* g2  = (const float*)d_in[14];
  const float* be2 = (const float*)d_in[15];
  float* outp = (float*)d_out;

  char* w = (char*)d_ws;
  size_t off = 0;
  auto alloc = [&](size_t bytes) -> void* {
    void* p = (void*)(w + off);
    off += (bytes + 255) & ~(size_t)255;
    return p;
  };
  __bf16* x_bf = (__bf16*)alloc((size_t)ROWS * DM * 2);
  __bf16* WqT  = (__bf16*)alloc((size_t)DM * DM * 2);
  __bf16* WkT  = (__bf16*)alloc((size_t)DM * DM * 2);
  __bf16* WvT  = (__bf16*)alloc((size_t)DM * DM * 2);
  __bf16* W1T  = (__bf16*)alloc((size_t)DM * FFN_DIM * 2);
  __bf16* W2T  = (__bf16*)alloc((size_t)FFN_DIM * DM * 2);
  __bf16* ErB  = (__bf16*)alloc((size_t)S_LEN * D_HEAD * 2);
  __bf16* qb   = (__bf16*)alloc((size_t)BATCH * N_HEAD * S_LEN * D_HEAD * 2);
  __bf16* kb   = (__bf16*)alloc((size_t)BATCH * N_HEAD * S_LEN * D_HEAD * 2);
  __bf16* vtb  = (__bf16*)alloc((size_t)BATCH * N_HEAD * S_LEN * D_HEAD * 2);
  float*  ctx  = (float*) alloc((size_t)ROWS * DM * 4);
  float*  h1   = (float*) alloc((size_t)ROWS * DM * 4);
  __bf16* h1b  = (__bf16*)alloc((size_t)ROWS * DM * 2);
  __bf16* y1   = (__bf16*)alloc((size_t)ROWS * FFN_DIM * 2);
  float*  y2   = (float*) alloc((size_t)ROWS * DM * 4);

  auto blocks = [](int n) { return (n + 255) / 256; };
  cvt_kernel<<<blocks(ROWS * DM),      256, 0, stream>>>(x,  x_bf, ROWS, DM, 0);
  cvt_kernel<<<blocks(DM * DM),        256, 0, stream>>>(Wq, WqT, DM, DM, 1);
  cvt_kernel<<<blocks(DM * DM),        256, 0, stream>>>(Wk, WkT, DM, DM, 1);
  cvt_kernel<<<blocks(DM * DM),        256, 0, stream>>>(Wv, WvT, DM, DM, 1);
  cvt_kernel<<<blocks(DM * FFN_DIM),   256, 0, stream>>>(W1, W1T, DM, FFN_DIM, 1);
  cvt_kernel<<<blocks(FFN_DIM * DM),   256, 0, stream>>>(W2, W2T, FFN_DIM, DM, 1);
  cvt_kernel<<<blocks(S_LEN * D_HEAD), 256, 0, stream>>>(Er, ErB, S_LEN, D_HEAD, 0);

  // QKV projections: tiles = (4096/32)*(512/64) = 1024 -> 256 blocks of 4 waves
  gemm_kernel<<<256,  128, 0, stream>>>(x_bf, WqT, bq, qb,  ROWS, DM, DM, 0);
  gemm_kernel<<<256,  128, 0, stream>>>(x_bf, WkT, bk, kb,  ROWS, DM, DM, 1);
  gemm_kernel<<<256,  128, 0, stream>>>(x_bf, WvT, bv, vtb, ROWS, DM, DM, 2);

  attn_kernel<<<dim3(S_LEN / 16 / 4, BATCH * N_HEAD), 128, 0, stream>>>(qb, kb, vtb, ErB, ctx);

  ln_kernel<<<ROWS, 256, 0, stream>>>(ctx, x, g1, be1, h1, h1b);

  // FFN1: (4096/32)*(2048/64) = 4096 tiles -> 1024 blocks
  gemm_kernel<<<1024, 128, 0, stream>>>(h1b, W1T, b1, y1, ROWS, FFN_DIM, DM, 3);
  // FFN2: (4096/32)*(512/64) = 1024 tiles -> 256 blocks
  gemm_kernel<<<256,  128, 0, stream>>>(y1,  W2T, b2, y2, ROWS, DM, FFN_DIM, 4);

  ln_kernel<<<ROWS, 256, 0, stream>>>(y2, h1, g2, be2, outp, nullptr);
}
